// RWKV4_AttentionSubLayer_18159121727563
// MI455X (gfx1250) — matrix operand order repaired and checked by reading, not compiled, not run on hardware
//
#include <hip/hip_runtime.h>
#include <hip/hip_bf16.h>

// ---------------------------------------------------------------------------
// RWKV4 attention sublayer for MI455X (gfx1250), wave32 + WMMA.
// B=4, T=1024, C=1024.  Pipeline:
//   1) mix+cvt   : xm_{k,v,r} = bf16(x*tm + shift(x)*(1-tm))
//   2) wcvt      : bf16 copies of Wk/Wv/Wr/Wo
//   3) gemm x3   : k,v,r = xm @ W^T   (v_wmma_f32_16x16x32_bf16, fp32 acc)
//   4) scan x3   : chunk-parallel WKV recurrence (16 chunks of 64 steps)
//                  + sigmoid gate, fp32 math, -> bf16 y
//   5) gemm      : d_out = y @ Wo^T   (fp32 out)
// ---------------------------------------------------------------------------

#define RB 4
#define RT 1024
#define RC 1024
#define GK 1024   // GEMM K dim (=C)
#define GN 1024   // GEMM N dim (=C)
#define GM 4096   // GEMM M dim (=B*T)

#define NCH 16            // scan chunks along T
#define CHL (RT / NCH)    // 64 steps per chunk

typedef __attribute__((ext_vector_type(16))) __bf16 v16bf;
typedef __attribute__((ext_vector_type(8)))  float  v8f;

union Frag {
  v16bf v;
  uint4 q[2];
};

__device__ __forceinline__ unsigned short f32_to_bf16_rne(float f) {
  unsigned int u = __float_as_uint(f);
  u += 0x7FFFu + ((u >> 16) & 1u);   // round to nearest even
  return (unsigned short)(u >> 16);
}

// ---------------------------------------------------------------------------
// 1) time-shift mix + bf16 conversion for the three GEMM A-operands
// ---------------------------------------------------------------------------
__global__ void __launch_bounds__(256)
mix_cvt_kernel(const float* __restrict__ x,
               const float* __restrict__ tmk,
               const float* __restrict__ tmv,
               const float* __restrict__ tmr,
               unsigned short* __restrict__ xmk,
               unsigned short* __restrict__ xmv,
               unsigned short* __restrict__ xmr) {
  int idx = blockIdx.x * 256 + threadIdx.x;          // [0, B*T*C)
  int c = idx & (RC - 1);
  int t = (idx >> 10) & (RT - 1);
  float xc = x[idx];
  float xp = (t > 0) ? x[idx - RC] : 0.0f;           // xx[t] = x[t-1], xx[0]=0
  float mk = tmk[c], mv = tmv[c], mr = tmr[c];
  xmk[idx] = f32_to_bf16_rne(xc * mk + xp * (1.0f - mk));
  xmv[idx] = f32_to_bf16_rne(xc * mv + xp * (1.0f - mv));
  xmr[idx] = f32_to_bf16_rne(xc * mr + xp * (1.0f - mr));
}

// ---------------------------------------------------------------------------
// 2) weight conversion to bf16
// ---------------------------------------------------------------------------
__global__ void __launch_bounds__(256)
wcvt_kernel(const float* __restrict__ Wk, const float* __restrict__ Wv,
            const float* __restrict__ Wr, const float* __restrict__ Wo,
            unsigned short* __restrict__ wkb, unsigned short* __restrict__ wvb,
            unsigned short* __restrict__ wrb, unsigned short* __restrict__ wob) {
  int idx = blockIdx.x * 256 + threadIdx.x;          // [0, C*C)
  wkb[idx] = f32_to_bf16_rne(Wk[idx]);
  wvb[idx] = f32_to_bf16_rne(Wv[idx]);
  wrb[idx] = f32_to_bf16_rne(Wr[idx]);
  wob[idx] = f32_to_bf16_rne(Wo[idx]);
}

// ---------------------------------------------------------------------------
// 3/5) bf16 WMMA GEMM:  C[M,N] = A[M,K] * W[N,K]^T, fp32 accumulate.
// Block = 256 threads = 8 waves, arranged 4 (M) x 2 (N).
// Wave tile = 32x64 (2x4 of 16x16), block tile = 128x128.
// A fragment layout (16x32 bf16, ISA 7.12.2): lane half h (=lane>>4), row m=lane&15
//   holds K = [8h..8h+7] and [16+8h..16+8h+7] -> two 16B contiguous loads.
// B fragment (32x16): lane n=lane&15 of half h holds K = [8h..8h+7] and
//   [16+8h..16+8h+7] of column n; since B = W^T these come from row (nBase+n) of W.
// ---------------------------------------------------------------------------
__global__ void __launch_bounds__(256)
gemm_bf16_nt(const unsigned short* __restrict__ A,   // [M,K] bf16 bits
             const unsigned short* __restrict__ W,   // [N,K] bf16 bits
             float* __restrict__ C) {                // [M,N] fp32
  const int tid  = threadIdx.x;
  const int lane = tid & 31;
  const int wave = tid >> 5;
  const int wm   = wave & 3;            // 4 waves along M
  const int wn   = wave >> 2;           // 2 waves along N
  const int l15  = lane & 15;
  const int h    = lane >> 4;

  const int mBase = blockIdx.y * 128 + wm * 32;
  const int nBase = blockIdx.x * 128 + wn * 64;

  const unsigned short* aRow0 = A + (size_t)(mBase + l15) * GK;
  const unsigned short* aRow1 = A + (size_t)(mBase + 16 + l15) * GK;
  const unsigned short* bRow[4];
#pragma unroll
  for (int j = 0; j < 4; ++j)
    bRow[j] = W + (size_t)(nBase + j * 16 + l15) * GK;

  v8f acc[2][4] = {};

  for (int k0 = 0; k0 < GK; k0 += 32) {
    Frag a[2], b[4];
    {
      const unsigned short* p0 = aRow0 + k0 + 8 * h;
      a[0].q[0] = *(const uint4*)(p0);
      a[0].q[1] = *(const uint4*)(p0 + 16);
      const unsigned short* p1 = aRow1 + k0 + 8 * h;
      a[1].q[0] = *(const uint4*)(p1);
      a[1].q[1] = *(const uint4*)(p1 + 16);
    }
#pragma unroll
    for (int j = 0; j < 4; ++j) {
      const unsigned short* p = bRow[j] + k0 + 8 * h;
      b[j].q[0] = *(const uint4*)(p);
      b[j].q[1] = *(const uint4*)(p + 16);
    }
#pragma unroll
    for (int s = 0; s < 2; ++s)
#pragma unroll
      for (int j = 0; j < 4; ++j)
        acc[s][j] = __builtin_amdgcn_wmma_f32_16x16x32_bf16(
            false, a[s].v, false, b[j].v, (short)0, acc[s][j], false, false);
  }

  // C/D layout: VGPR r -> row (h*8 + r), col = lane&15
#pragma unroll
  for (int s = 0; s < 2; ++s) {
    const int mrow = mBase + s * 16 + h * 8;
#pragma unroll
    for (int j = 0; j < 4; ++j) {
      const int ncol = nBase + j * 16 + l15;
      float* cp = C + (size_t)mrow * GN + ncol;
#pragma unroll
      for (int rr = 0; rr < 8; ++rr)
        cp[(size_t)rr * GN] = acc[s][j][rr];
    }
  }
}

// ---------------------------------------------------------------------------
// 4) Chunk-parallel WKV recurrence.
//    wkv[t] = e^{tf}*kv[t] + a_t ;  a_{t+1} = w*a_t + kv[t],  w = e^{-e^{td}}
//    Linear state -> blocked scan:
//      pass1: per (b,c,chunk) local sums S = running recurrence from 0
//      pass2: per (b,c) combine:  A_{j+1} = w^L * A_j + S_j  (store A_j)
//      pass3: per (b,c,chunk) replay with seeded state, gate, emit bf16 y
// Thread mapping (pass1/3): c = idx&1023 (fastest, coalesced), j = chunk.
// ---------------------------------------------------------------------------
__global__ void __launch_bounds__(256)
wkv_pass1_kernel(const float* __restrict__ kf, const float* __restrict__ vf,
                 const float* __restrict__ time_decay,
                 float* __restrict__ Skv, float* __restrict__ Sk) {
  int idx = blockIdx.x * 256 + threadIdx.x;   // [0, B*NCH*C)
  int c = idx & (RC - 1);
  int j = (idx >> 10) & (NCH - 1);
  int b = idx >> 14;
  float wstep = __expf(-__expf(time_decay[c]));
  float sKV = 0.0f, sK = 0.0f;
  size_t base = ((size_t)b * RT + (size_t)j * CHL) * RC + c;
  for (int t = 0; t < CHL; ++t) {
    size_t o = base + (size_t)t * RC;
    float kc = __expf(fminf(kf[o], 60.0f));
    float kv = kc * vf[o];
    sKV = wstep * sKV + kv;
    sK  = wstep * sK  + kc;
  }
  Skv[idx] = sKV;
  Sk[idx]  = sK;
}

__global__ void __launch_bounds__(256)
wkv_pass2_kernel(const float* __restrict__ Skv, const float* __restrict__ Sk,
                 const float* __restrict__ time_decay,
                 float* __restrict__ Akv, float* __restrict__ Ak) {
  int idx = blockIdx.x * 256 + threadIdx.x;   // [0, B*C)
  int c = idx & (RC - 1);
  int b = idx >> 10;
  float decay = __expf(time_decay[c]);
  float wL = __expf(-decay * (float)CHL);     // w^L
  float aKV = 0.0f, aK = 0.0f;
#pragma unroll
  for (int j = 0; j < NCH; ++j) {
    size_t o = ((size_t)b * NCH + j) * RC + c;
    Akv[o] = aKV;                             // state entering chunk j
    Ak[o]  = aK;
    aKV = wL * aKV + Skv[o];
    aK  = wL * aK  + Sk[o];
  }
}

__global__ void __launch_bounds__(256)
wkv_pass3_kernel(const float* __restrict__ kf, const float* __restrict__ vf,
                 const float* __restrict__ rf,
                 const float* __restrict__ time_decay,
                 const float* __restrict__ time_first,
                 const float* __restrict__ Akv, const float* __restrict__ Ak,
                 unsigned short* __restrict__ ymid) {
  int idx = blockIdx.x * 256 + threadIdx.x;   // [0, B*NCH*C)
  int c = idx & (RC - 1);
  int j = (idx >> 10) & (NCH - 1);
  int b = idx >> 14;
  float wstep = __expf(-__expf(time_decay[c]));
  float ef    = __expf(time_first[c]);
  float aKV = Akv[idx];
  float aK  = Ak[idx];
  size_t base = ((size_t)b * RT + (size_t)j * CHL) * RC + c;
  for (int t = 0; t < CHL; ++t) {
    size_t o = base + (size_t)t * RC;
    float kc = __expf(fminf(kf[o], 60.0f));   // RWKV_K_CLAMP
    float kv = kc * vf[o];
    float wkv = ef * kv + aKV;
    float wk  = ef * kc + aK + 1e-8f;         // RWKV_K_EPS
    float sr  = 1.0f / (1.0f + __expf(-rf[o]));
    ymid[o] = f32_to_bf16_rne(sr * (wkv / wk));
    aKV = wstep * aKV + kv;
    aK  = wstep * aK  + kc;
  }
}

// ---------------------------------------------------------------------------
extern "C" void kernel_launch(void* const* d_in, const int* in_sizes, int n_in,
                              void* d_out, int out_size, void* d_ws, size_t ws_size,
                              hipStream_t stream) {
  const float* xq         = (const float*)d_in[0];
  // d_in[1] (xk), d_in[2] (xv) unused by the reference module
  const float* time_decay = (const float*)d_in[3];
  const float* time_first = (const float*)d_in[4];
  const float* tmk        = (const float*)d_in[5];
  const float* tmv        = (const float*)d_in[6];
  const float* tmr        = (const float*)d_in[7];
  const float* Wk         = (const float*)d_in[8];
  const float* Wv         = (const float*)d_in[9];
  const float* Wr         = (const float*)d_in[10];
  const float* Wo         = (const float*)d_in[11];

  char* ws = (char*)d_ws;
  const size_t MB = 1024u * 1024u;
  unsigned short* xmk = (unsigned short*)(ws + 0 * MB);   // 8 MB
  unsigned short* xmv = (unsigned short*)(ws + 8 * MB);   // 8 MB
  unsigned short* xmr = (unsigned short*)(ws + 16 * MB);  // 8 MB
  unsigned short* wkb = (unsigned short*)(ws + 24 * MB);  // 2 MB
  unsigned short* wvb = (unsigned short*)(ws + 26 * MB);  // 2 MB
  unsigned short* wrb = (unsigned short*)(ws + 28 * MB);  // 2 MB
  unsigned short* wob = (unsigned short*)(ws + 30 * MB);  // 2 MB
  float*          kf  = (float*)(ws + 32 * MB);           // 16 MB
  float*          vf  = (float*)(ws + 48 * MB);           // 16 MB
  float*          rf  = (float*)(ws + 64 * MB);           // 16 MB
  float*          Skv = (float*)(ws + 80 * MB);           // 256 KB
  float*          Sk  = (float*)(ws + 80 * MB + 256 * 1024);
  float*          Akv = (float*)(ws + 80 * MB + 512 * 1024);
  float*          Ak  = (float*)(ws + 80 * MB + 768 * 1024);
  unsigned short* ymid = xmk;  // reuse xmk after the k-GEMM consumed it

  const int elems = RB * RT * RC;                 // 4M
  mix_cvt_kernel<<<elems / 256, 256, 0, stream>>>(xq, tmk, tmv, tmr, xmk, xmv, xmr);
  wcvt_kernel<<<(RC * RC) / 256, 256, 0, stream>>>(Wk, Wv, Wr, Wo, wkb, wvb, wrb, wob);

  dim3 ggrid(GN / 128, GM / 128);                 // (8, 32)
  gemm_bf16_nt<<<ggrid, 256, 0, stream>>>(xmk, wkb, kf);
  gemm_bf16_nt<<<ggrid, 256, 0, stream>>>(xmv, wvb, vf);
  gemm_bf16_nt<<<ggrid, 256, 0, stream>>>(xmr, wrb, rf);

  const int scanThreads = RB * NCH * RC;          // 65536
  wkv_pass1_kernel<<<scanThreads / 256, 256, 0, stream>>>(kf, vf, time_decay, Skv, Sk);
  wkv_pass2_kernel<<<(RB * RC) / 256, 256, 0, stream>>>(Skv, Sk, time_decay, Akv, Ak);
  wkv_pass3_kernel<<<scanThreads / 256, 256, 0, stream>>>(kf, vf, rf, time_decay,
                                                          time_first, Akv, Ak, ymid);

  gemm_bf16_nt<<<ggrid, 256, 0, stream>>>(ymid, wob, (float*)d_out);
}
